// CSNNPC_55611236549422
// MI455X (gfx1250) — compile-verified, matrix operand order checked by reading, not run on hardware
//
#include <hip/hip_runtime.h>

typedef _Float16 v8h  __attribute__((ext_vector_type(8)));
typedef _Float16 v16h __attribute__((ext_vector_type(16)));
typedef float    v8f  __attribute__((ext_vector_type(8)));

#define NSTEP 25
#define BTOT  4096
#define POUT  256
#define BP    (BTOT * POUT)            // 1048576
#define TBP   ((size_t)NSTEP * BP)     // 26214400

// LDS row strides (halves), multiples of 8 (16B aligned), tuned for bank spread
#define S1 488   // spk1: 15 pos * 32 ch = 480 used
#define S2 904   // spk2: 7 pos * 128 ch = 896 used
#define S3 776   // spk3: 256 ch * 3 pos = 768 used

// packed-weight offsets in d_ws (halves)
#define PK_W2 0
#define PK_W3 8192
#define PK_WF 73728
#define PK_TOT 270336

// 16-bit WMMA A/B fragment K-mapping (ISA 7.12.2): lane holds K in two 8-half runs
__device__ __forceinline__ int kmap(int l, int j) {
    int hi = (l >> 4) * 8;
    return (j < 8) ? (hi + j) : (16 + hi + (j - 8));
}

// ---------------------------------------------------------------------------
// Prepack: f32 weights -> f16 B-matrix fragments (lane-major, 2x16B per lane)
// ---------------------------------------------------------------------------
__global__ void snn_prepack(const float* __restrict__ w2, const float* __restrict__ w3,
                            const float* __restrict__ wf, _Float16* __restrict__ pk) {
    int id = blockIdx.x * blockDim.x + threadIdx.x;
    if (id >= PK_TOT) return;
    float v;
    if (id < PK_W3) {                 // w2: K=64 (2 chunks) x N=128 (8 tiles)
        int rel = id, frag = rel >> 9, r2 = rel & 511, l = r2 >> 4, j = r2 & 15;
        int kc = frag >> 3, nt = frag & 7;
        int k = kc * 32 + kmap(l, j);
        int o = nt * 16 + (l & 15);
        v = w2[(o * 32 + (k & 31)) * 2 + (k >> 5)];
    } else if (id < PK_WF) {          // w3: K=256 (8 chunks) x N=256 (16 tiles)
        int rel = id - PK_W3, frag = rel >> 9, r2 = rel & 511, l = r2 >> 4, j = r2 & 15;
        int kc = frag >> 4, nt = frag & 15;
        int k = kc * 32 + kmap(l, j);
        int o = nt * 16 + (l & 15);
        v = w3[(o * 128 + (k & 127)) * 2 + (k >> 7)];
    } else {                          // wf: K=768 (24 chunks) x N=256 (16 tiles)
        int rel = id - PK_WF, frag = rel >> 9, r2 = rel & 511, l = r2 >> 4, j = r2 & 15;
        int kc = frag >> 4, nt = frag & 15;
        int k = kc * 32 + kmap(l, j);
        int p = nt * 16 + (l & 15);
        v = wf[p * 768 + k];
    }
    pk[id] = (_Float16)v;
}

// ---------------------------------------------------------------------------
// Fragment helpers
// ---------------------------------------------------------------------------
__device__ __forceinline__ v16h load_a(const _Float16* base, int strideH, int kbase, int lane) {
    int m = lane & 15, hi = (lane >> 4) * 8;
    const v8h* p0 = (const v8h*)(base + m * strideH + kbase + hi);
    const v8h* p1 = (const v8h*)(base + m * strideH + kbase + 16 + hi);
    v8h lo = *p0, hv = *p1;
    return __builtin_shufflevector(lo, hv, 0,1,2,3,4,5,6,7,8,9,10,11,12,13,14,15);
}

__device__ __forceinline__ v16h load_b(const _Float16* pk, int frag, int lane) {
    const v8h* p = (const v8h*)(pk + ((size_t)frag * 32 + lane) * 16);
    v8h lo = p[0], hv = p[1];
    return __builtin_shufflevector(lo, hv, 0,1,2,3,4,5,6,7,8,9,10,11,12,13,14,15);
}

__device__ __forceinline__ v8f wmma16(v16h a, v16h b, v8f c) {
    return __builtin_amdgcn_wmma_f32_16x16x32_f16(false, a, false, b, (short)0, c, false, false);
}

// reset-by-subtraction LIF on an accumulator fragment; updates mem, returns spikes
__device__ __forceinline__ v8f lif_frag(v8f cur, v8f& mem, float beta, float thr) {
    v8f spk;
#pragma unroll
    for (int r = 0; r < 8; ++r) {
        float m  = mem[r];
        float mn = beta * m + cur[r] - ((m > thr) ? thr : 0.0f);
        mem[r] = mn;
        spk[r] = (mn > thr) ? 1.0f : 0.0f;
    }
    return spk;
}

// ---------------------------------------------------------------------------
// Persistent SNN kernel: 256 blocks x 256 threads, 16 batch rows per block.
// __launch_bounds__(256, 4): cap at 4 waves/SIMD -> <=256 VGPRs/wave, so the
// two 128KB-LDS workgroups per WGP can actually co-reside and no VGPR-MSB
// switching is needed.
// ---------------------------------------------------------------------------
__global__ __launch_bounds__(256, 4)
void snn_main(const float* __restrict__ x,  const float* __restrict__ w1,
              const float* __restrict__ b1, const float* __restrict__ b2,
              const float* __restrict__ b3, const float* __restrict__ bf,
              const float* __restrict__ betas, const float* __restrict__ thrs,
              const _Float16* __restrict__ pk, float* __restrict__ out) {
    extern __shared__ char smem[];
    float*    cur1L = (float*)smem;                 // [16][480]
    float*    mem1L = cur1L + 7680;                 // [16][480]
    _Float16* spk1L = (_Float16*)(mem1L + 7680);    // [16][S1]
    _Float16* spk2L = spk1L + 16 * S1;              // [16][S2]
    _Float16* spk3L = spk2L + 16 * S2;              // [16][S3]

    const int tid  = threadIdx.x;
    const int lane = tid & 31;
    const int wv   = tid >> 5;
    const int row0 = blockIdx.x * 16;
    const int ncol = lane & 15;          // C-frag column within tile (ISA layout)
    const int rhi  = (lane >> 4) * 8;    // row offset for lanes 16..31

    float be[4], th[4];
#pragma unroll
    for (int i = 0; i < 4; ++i) {
        be[i] = fminf(fmaxf(betas[i], 0.0f), 1.0f);
        th[i] = thrs[i];
    }

    // loop-invariant cur1 = maxpool2(conv1(x)) in [pos*32+ch] layout; zero mem1
    for (int idx = tid; idx < 7680; idx += 256) {
        int row = idx / 480, c = idx % 480;
        int p = c >> 5, ch = c & 31;
        const float* xr = x + (size_t)(row0 + row) * 32;
        float wa = w1[ch * 2], wb = w1[ch * 2 + 1], bb = b1[ch];
        float x0 = xr[2 * p], x1 = xr[2 * p + 1], x2 = xr[2 * p + 2];
        cur1L[idx] = fmaxf(wa * x0 + wb * x1 + bb, wa * x1 + wb * x2 + bb);
        mem1L[idx] = 0.0f;
    }

    // per-wave loop-invariant weight fragments (w2) and biases
    v16h Bw2_0 = load_b(pk + PK_W2, 0 * 8 + wv, lane);
    v16h Bw2_1 = load_b(pk + PK_W2, 1 * 8 + wv, lane);
    const float bias2  = b2[wv * 16 + ncol];
    const float bias3a = b3[wv * 16 + ncol],        bias3b = b3[(wv + 8) * 16 + ncol];
    const float biasfa = bf[wv * 16 + ncol],        biasfb = bf[(wv + 8) * 16 + ncol];

    v8f mem2[7], mem3[6], mem4[2];
#pragma unroll
    for (int i = 0; i < 7; ++i) mem2[i] = (v8f){};
#pragma unroll
    for (int i = 0; i < 6; ++i) mem3[i] = (v8f){};
    mem4[0] = (v8f){}; mem4[1] = (v8f){};

    __syncthreads();

    for (int t = 0; t < NSTEP; ++t) {
        // ---- phase 1: LIF1 elementwise, emit spk1 (f16, A-layout rows) ----
        for (int idx = tid; idx < 7680; idx += 256) {
            int row = idx / 480, c = idx % 480;
            float m  = mem1L[idx];
            float mn = be[0] * m + cur1L[idx] - ((m > th[0]) ? th[0] : 0.0f);
            mem1L[idx] = mn;
            spk1L[row * S1 + c] = (_Float16)((mn > th[0]) ? 1.0f : 0.0f);
        }
        __syncthreads();

        // ---- phase 2: conv2 + maxpool + LIF2 (wave owns N-tile wv) ----
        for (int p = 0; p < 7; ++p) {
            v16h A0 = load_a(spk1L, S1, (2 * p) * 32, lane);
            v16h A1 = load_a(spk1L, S1, (2 * p + 1) * 32, lane);
            v16h A2 = load_a(spk1L, S1, (2 * p + 2) * 32, lane);
            v8f accA = {}, accB = {};
            accA = wmma16(A0, Bw2_0, accA);
            accA = wmma16(A1, Bw2_1, accA);
            accB = wmma16(A1, Bw2_0, accB);
            accB = wmma16(A2, Bw2_1, accB);
            v8f cur;
#pragma unroll
            for (int r = 0; r < 8; ++r) cur[r] = fmaxf(accA[r], accB[r]) + bias2;
            v8f spk = lif_frag(cur, mem2[p], be[1], th[1]);
            int col = p * 128 + wv * 16 + ncol;
#pragma unroll
            for (int r = 0; r < 8; ++r)
                spk2L[(r + rhi) * S2 + col] = (_Float16)spk[r];
        }
        __syncthreads();

        // ---- phase 3: conv3 + maxpool + LIF3 (wave owns N-tiles wv, wv+8) ----
        for (int ni = 0; ni < 2; ++ni) {
            int nt = wv + ni * 8;
            float bias = ni ? bias3b : bias3a;
            for (int p = 0; p < 3; ++p) {
                v8f accA = {}, accB = {};
#pragma unroll 2
                for (int kc = 0; kc < 8; ++kc) {
                    v16h Bf = load_b(pk + PK_W3, kc * 16 + nt, lane);
                    v16h Aa = load_a(spk2L, S2, (8 * p + kc) * 32, lane);
                    accA = wmma16(Aa, Bf, accA);
                    v16h Ab = load_a(spk2L, S2, (8 * p + 4 + kc) * 32, lane);
                    accB = wmma16(Ab, Bf, accB);
                }
                v8f cur;
#pragma unroll
                for (int r = 0; r < 8; ++r) cur[r] = fmaxf(accA[r], accB[r]) + bias;
                v8f spk = lif_frag(cur, mem3[ni * 3 + p], be[2], th[2]);
                int col = (nt * 16 + ncol) * 3 + p;   // fc flatten: ch*3 + pos
#pragma unroll
                for (int r = 0; r < 8; ++r)
                    spk3L[(r + rhi) * S3 + col] = (_Float16)spk[r];
            }
        }
        __syncthreads();

        // ---- phase 4: FC [16x768]@[768x256] + LIF4 + record outputs ----
        for (int ni = 0; ni < 2; ++ni) {
            int nt = wv + ni * 8;
            v8f acc = {};
#pragma unroll 2
            for (int kc = 0; kc < 24; ++kc) {
                if (kc < 23)
                    __builtin_prefetch(pk + PK_WF + ((size_t)((kc + 1) * 16 + nt) * 32 + lane) * 16, 0, 1);
                v16h Bf = load_b(pk + PK_WF, kc * 16 + nt, lane);
                v16h Af = load_a(spk3L, S3, kc * 32, lane);
                acc = wmma16(Af, Bf, acc);
            }
            float bias = ni ? biasfb : biasfa;
            v8f cur;
#pragma unroll
            for (int r = 0; r < 8; ++r) cur[r] = acc[r] + bias;
            v8f spk = lif_frag(cur, mem4[ni], be[3], th[3]);
            int n = nt * 16 + ncol;
            size_t tb = (size_t)t * BP;
#pragma unroll
            for (int r = 0; r < 8; ++r) {
                int rowg = row0 + r + rhi;
                size_t off = tb + (size_t)rowg * POUT + n;
                out[off]           = cur[r];        // cur_rec
                out[TBP + off]     = spk[r];        // spk_rec
                out[2 * TBP + off] = mem4[ni][r];   // mem_rec
            }
        }
        __syncthreads();
    }
}

// ---------------------------------------------------------------------------
extern "C" void kernel_launch(void* const* d_in, const int* in_sizes, int n_in,
                              void* d_out, int out_size, void* d_ws, size_t ws_size,
                              hipStream_t stream) {
    (void)in_sizes; (void)n_in; (void)out_size; (void)ws_size;
    const float* x  = (const float*)d_in[0];
    const float* w1 = (const float*)d_in[1];
    const float* b1 = (const float*)d_in[2];
    const float* w2 = (const float*)d_in[3];
    const float* b2 = (const float*)d_in[4];
    const float* w3 = (const float*)d_in[5];
    const float* b3 = (const float*)d_in[6];
    const float* wf = (const float*)d_in[7];
    const float* bf = (const float*)d_in[8];
    const float* betas = (const float*)d_in[9];
    const float* thrs  = (const float*)d_in[10];
    _Float16* pk = (_Float16*)d_ws;

    snn_prepack<<<(PK_TOT + 255) / 256, 256, 0, stream>>>(w2, w3, wf, pk);

    size_t smem = 7680 * 4 * 2 + (size_t)(16 * S1 + 16 * S2 + 16 * S3) * 2;
    snn_main<<<BTOT / 16, 256, smem, stream>>>(x, w1, b1, b2, b3, bf, betas, thrs,
                                               pk, (float*)d_out);
}
